// LSTMCellWithResidual_86706799771972
// MI455X (gfx1250) — compile-verified, matrix operand order checked by reading, not run on hardware
//
#include <hip/hip_runtime.h>
#include <cstdint>
#include <cstddef>

// ---------------------------------------------------------------------------
// LSTM cell + residual, fused bf16-WMMA implementation for gfx1250 (MI455X).
// Compute-bound (~447 GFLOP vs ~400MB HBM traffic) -> v_wmma_f32_16x16x32_bf16.
// A/B tiles staged into double-buffered LDS by the Tensor Data Mover
// (tensor_load_to_lds pipelined against WMMA via s_wait_tensorcnt 2).
// Elementwise LSTM math fused into 5 specialized K-reduction passes per tile.
// ---------------------------------------------------------------------------

#define BATCH 8192
#define DIN   1024
#define DH    2048
#define DK    (DIN + DH)   // 3072

typedef __attribute__((ext_vector_type(16))) __bf16        v16bf;
typedef __attribute__((ext_vector_type(8)))  __bf16        v8bf;
typedef __attribute__((ext_vector_type(8)))  float         v8f;
typedef __attribute__((ext_vector_type(4)))  unsigned int  u32x4;
typedef __attribute__((ext_vector_type(8)))  int           i32x8;
typedef __attribute__((ext_vector_type(4)))  int           i32x4;

#if __has_builtin(__builtin_amdgcn_tensor_load_to_lds) && \
    __has_builtin(__builtin_amdgcn_s_wait_tensorcnt)
#define USE_TDM 1
#else
#define USE_TDM 0
#endif

#if __has_builtin(__builtin_amdgcn_tanhf)
#define TANHF(x) __builtin_amdgcn_tanhf(x)
#else
#define TANHF(x) tanhf(x)
#endif

__device__ __forceinline__ unsigned short f2bf(float f) {
    union { float f; unsigned u; } v; v.f = f;
    unsigned r = v.u + 0x7fffu + ((v.u >> 16) & 1u);   // round-to-nearest-even
    return (unsigned short)(r >> 16);
}

__device__ __forceinline__ float sigmoidf(float x) {
    return 1.0f / (1.0f + __expf(-x));
}

__device__ __forceinline__ v16bf cat8(v8bf lo, v8bf hi) {
    return __builtin_shufflevector(lo, hi, 0,1,2,3,4,5,6,7,8,9,10,11,12,13,14,15);
}

// ---------------------------------------------------------------------------
// Conversion kernels (one-time, memory bound): fp32 -> bf16, weights
// pre-transposed to [out][k] so GEMM B-tiles are contiguous row-major slabs.
// ---------------------------------------------------------------------------

__global__ __launch_bounds__(256) void cvt_xh_kernel(
    const float* __restrict__ x, const float* __restrict__ h,
    unsigned short* __restrict__ xh)
{
    size_t idx = (size_t)blockIdx.x * blockDim.x + threadIdx.x;
    if (idx >= (size_t)BATCH * DK) return;
    int row = (int)(idx / DK);
    int col = (int)(idx % DK);
    float v = (col < DIN) ? x[(size_t)row * DIN + col]
                          : h[(size_t)row * DH + (col - DIN)];
    xh[idx] = f2bf(v);
}

// W_gates [4][DK][DH] -> wgT [4][DH][DK] bf16
__global__ __launch_bounds__(256) void cvt_wgT_kernel(
    const float* __restrict__ w, unsigned short* __restrict__ o)
{
    size_t idx = (size_t)blockIdx.x * blockDim.x + threadIdx.x;
    if (idx >= (size_t)4 * DH * DK) return;
    int g = (int)(idx / ((size_t)DH * DK));
    size_t r = idx % ((size_t)DH * DK);
    int n = (int)(r / DK);
    int k = (int)(r % DK);
    o[idx] = f2bf(w[(size_t)g * DK * DH + (size_t)k * DH + n]);
}

// W_res [DIN][DH] -> wrT [DH][DIN] bf16
__global__ __launch_bounds__(256) void cvt_wrT_kernel(
    const float* __restrict__ w, unsigned short* __restrict__ o)
{
    size_t idx = (size_t)blockIdx.x * blockDim.x + threadIdx.x;
    if (idx >= (size_t)DH * DIN) return;
    int n = (int)(idx / DIN);
    int k = (int)(idx % DIN);
    o[idx] = f2bf(w[(size_t)k * DH + n]);
}

// ---------------------------------------------------------------------------
// TDM helper: 2D tile load Global->LDS, 2-byte elements, with LDS padding of
// +4 DWORDs every 16 DWORDs (64B tile row -> 80B padded LDS pitch).
// ---------------------------------------------------------------------------
#if USE_TDM
__device__ __forceinline__ void tdm_load_2d(
    unsigned long long gaddr, unsigned lds_off,
    unsigned tensor_w, unsigned tensor_h,
    unsigned tile_w, unsigned tile_h,
    unsigned long long row_stride)   // in elements
{
    u32x4 g0;
    g0[0] = 1u;                                              // count=1 (valid)
    g0[1] = lds_off;                                         // lds_addr
    g0[2] = (unsigned)(gaddr & 0xffffffffull);               // global_addr lo
    g0[3] = (unsigned)((gaddr >> 32) & 0x01ffffffull)        // global_addr hi
          | (2u << 30);                                      // type = 2 (image)

    i32x8 g1;
    g1[0] = (int)((1u << 16)      // data_size = 1 (2 bytes)
                | (1u << 20)      // pad_enable
                | (3u << 22)      // pad_interval: 16 DWORDs
                | (3u << 25));    // pad_amount:   4 DWORDs
    g1[1] = (int)((tensor_w & 0xffffu) << 16);               // tensor_dim0 lo
    g1[2] = (int)((tensor_w >> 16) | ((tensor_h & 0xffffu) << 16));
    g1[3] = (int)((tensor_h >> 16) | (tile_w << 16));        // tile_dim0
    g1[4] = (int)(tile_h & 0xffffu);                         // tile_dim1, tile_dim2=0
    g1[5] = (int)(unsigned)(row_stride & 0xffffffffull);     // dim0_stride lo
    g1[6] = (int)(unsigned)((row_stride >> 32) & 0xffffull); // dim0_stride hi
    g1[7] = 0;                                               // dim1_stride (2D)

    i32x4 gz = (i32x4){0, 0, 0, 0};
#if __has_include(<hip/amd_detail/amd_gfx1250_TDM.h>)
    i32x8 gz8 = (i32x8){0, 0, 0, 0, 0, 0, 0, 0};
    __builtin_amdgcn_tensor_load_to_lds(g0, g1, gz, gz, gz8, 0);
#else
    __builtin_amdgcn_tensor_load_to_lds(g0, g1, gz, gz, 0);
#endif
}
#endif

// ---------------------------------------------------------------------------
// Fused 5-pass WMMA GEMM + LSTM epilogue.
//   Block: 256 threads = 8 waves; 128(B) x 64(H) output tile.
//   Waves: 4 along M (32 rows each) x 2 along N (32 cols each).
//   Wave sub-tile 32x32 = 2x2 fragments -> 32 acc + 32 carry VGPRs/lane.
//   Double-buffered LDS tiles; TDM load of tile i+1 overlaps WMMA on tile i.
// ---------------------------------------------------------------------------

#define TM 128
#define TN 64
#define TK 32
#define LDT 40   // padded LDS pitch in elements (80B rows, 16B aligned)

__global__ __launch_bounds__(256) void lstm_fused_wmma_kernel(
    const unsigned short* __restrict__ xh,      // [BATCH][DK] bf16
    const unsigned short* __restrict__ wgT,     // [4][DH][DK] bf16 (transposed)
    const unsigned short* __restrict__ wrT,     // [DH][DIN] bf16 (transposed)
    const float* __restrict__ b_gates,          // [4][DH]
    const float* __restrict__ b_res,            // [DH]
    const float* __restrict__ c_prev,           // [BATCH][DH]
    float* __restrict__ h_out,                  // [BATCH][DH]
    float* __restrict__ c_out)                  // [BATCH][DH]
{
    __shared__ __align__(16) unsigned short As[2][TM][LDT];  // xh tiles (db)
    __shared__ __align__(16) unsigned short Bt[2][TN][LDT];  // W tiles (db)

    const int tid  = threadIdx.x;
    const int lane = tid & 31;
    const int wave = tid >> 5;
    const int wm   = wave & 3;          // 4 waves along M
    const int wn   = wave >> 2;         // 2 waves along N
    const int lr   = lane & 15;
    const int half = lane >> 4;

    const int n0 = blockIdx.x * TN;
    const int m0 = blockIdx.y * TM;

    float carry[2][2][8];

#if USE_TDM
    const unsigned ldsA0 = (unsigned)(uintptr_t)&As[0][0][0];
    const unsigned ldsA1 = (unsigned)(uintptr_t)&As[1][0][0];
    const unsigned ldsB0 = (unsigned)(uintptr_t)&Bt[0][0][0];
    const unsigned ldsB1 = (unsigned)(uintptr_t)&Bt[1][0][0];
#endif

    // Pass order: i, chat, f, o, residual.  Fully unrolled -> each pass's
    // epilogue and operand bases specialize to constants.
    #pragma unroll
    for (int pass = 0; pass < 5; ++pass) {
        const bool is_res = (pass == 4);
        const int  g      = (pass == 0) ? 0 : (pass == 1) ? 3 : (pass == 2) ? 1 : 2;
        const int  Kext   = is_res ? DIN : DK;
        const int  ldB    = is_res ? DIN : DK;
        const unsigned short* __restrict__ Bsrc =
            is_res ? wrT : (wgT + (size_t)g * DH * DK);

        v8f acc[2][2];
        #pragma unroll
        for (int mf = 0; mf < 2; ++mf)
            #pragma unroll
            for (int nf = 0; nf < 2; ++nf)
                acc[mf][nf] = (v8f){0.f,0.f,0.f,0.f,0.f,0.f,0.f,0.f};

#if USE_TDM
        if (wave == 0) {
            // Prologue: fetch tile 0 into buffer 0.
            tdm_load_2d((unsigned long long)(uintptr_t)xh + ((size_t)m0 * DK) * 2ull,
                        ldsA0, DK, BATCH, TK, TM, DK);
            tdm_load_2d((unsigned long long)(uintptr_t)Bsrc + ((size_t)n0 * ldB) * 2ull,
                        ldsB0, (unsigned)ldB, DH, TK, TN, (unsigned long long)ldB);
        }
#endif
        int db = 0;
        #pragma unroll 1
        for (int k0 = 0; k0 < Kext; k0 += TK, db ^= 1) {
#if USE_TDM
            if (wave == 0) {
                if (k0 + TK < Kext) {
                    // Pipeline: issue tile i+1 into the other buffer, then wait
                    // until only those 2 loads remain in flight (tile i done).
                    const unsigned la = db ? ldsA0 : ldsA1;
                    const unsigned lb = db ? ldsB0 : ldsB1;
                    tdm_load_2d((unsigned long long)(uintptr_t)xh +
                                    ((size_t)m0 * DK + k0 + TK) * 2ull,
                                la, DK, BATCH, TK, TM, DK);
                    tdm_load_2d((unsigned long long)(uintptr_t)Bsrc +
                                    ((size_t)n0 * ldB + k0 + TK) * 2ull,
                                lb, (unsigned)ldB, DH, TK, TN,
                                (unsigned long long)ldB);
                    __builtin_amdgcn_s_wait_tensorcnt(2);
                } else {
                    __builtin_amdgcn_s_wait_tensorcnt(0);
                }
            }
#else
            // ---- manual staging fallback: A 512 chunks, B 256 chunks ----
            #pragma unroll
            for (int i = 0; i < 2; ++i) {
                int chunk = tid + i * 256;
                int row = chunk >> 2;
                int c8  = chunk & 3;
                const uint4 v = *(const uint4*)&xh[(size_t)(m0 + row) * DK + k0 + c8 * 8];
                *(uint4*)&As[db][row][c8 * 8] = v;
            }
            {
                int row = tid >> 2;
                int c8  = tid & 3;
                const uint4 v = *(const uint4*)&Bsrc[(size_t)(n0 + row) * ldB + k0 + c8 * 8];
                *(uint4*)&Bt[db][row][c8 * 8] = v;
            }
#endif
            __syncthreads();

            // ---- fragment loads (16-bit A/B striping) ----
            // lanes 0-15: K in {0..7,16..23}; lanes 16-31: K in {8..15,24..31}
            v16bf afrag[2];
            #pragma unroll
            for (int mf = 0; mf < 2; ++mf) {
                int r = wm * 32 + mf * 16 + lr;
                v8bf lo = *(const v8bf*)&As[db][r][half * 8];
                v8bf hi = *(const v8bf*)&As[db][r][16 + half * 8];
                afrag[mf] = cat8(lo, hi);
            }
            v16bf bfrag[2];
            #pragma unroll
            for (int nf = 0; nf < 2; ++nf) {
                int c = wn * 32 + nf * 16 + lr;
                v8bf lo = *(const v8bf*)&Bt[db][c][half * 8];
                v8bf hi = *(const v8bf*)&Bt[db][c][16 + half * 8];
                bfrag[nf] = cat8(lo, hi);
            }

            // ---- 4x v_wmma_f32_16x16x32_bf16 ----
            #pragma unroll
            for (int mf = 0; mf < 2; ++mf)
                #pragma unroll
                for (int nf = 0; nf < 2; ++nf)
                    acc[mf][nf] = __builtin_amdgcn_wmma_f32_16x16x32_bf16(
                        false, afrag[mf], false, bfrag[nf],
                        (short)0, acc[mf][nf], false, false);

            __syncthreads();
        }

        // ---- epilogue: fold pass result into carry / emit outputs ----
        // C/D layout: VGPR e, lanes 0-15 -> (M=e, N=lane); lanes 16-31 -> M=8+e.
        #pragma unroll
        for (int mf = 0; mf < 2; ++mf) {
            #pragma unroll
            for (int nf = 0; nf < 2; ++nf) {
                const int n = n0 + wn * 32 + nf * 16 + lr;
                const float bias = is_res ? b_res[n] : b_gates[g * DH + n];
                #pragma unroll
                for (int e = 0; e < 8; ++e) {
                    const int m = m0 + wm * 32 + mf * 16 + half * 8 + e;
                    const size_t off = (size_t)m * DH + n;
                    const float v = acc[mf][nf][e] + bias;
                    float cr = carry[mf][nf][e];
                    if (pass == 0) {                 // i-gate
                        cr = sigmoidf(v);
                    } else if (pass == 1) {          // chat
                        cr = cr * TANHF(v);
                    } else if (pass == 2) {          // f-gate -> c_t
                        float ct = sigmoidf(v) * c_prev[off] + cr;
                        c_out[off] = ct;
                        cr = TANHF(ct);
                    } else if (pass == 3) {          // o-gate
                        cr = cr * sigmoidf(v);
                    } else {                         // residual -> h_t
                        h_out[off] = cr + v;
                    }
                    carry[mf][nf][e] = cr;
                }
            }
        }
        __syncthreads();
    }
}

// ---------------------------------------------------------------------------
// Harness entry point.
// Inputs: x_t, h_t_1, c_t_1, W_gates, b_gates, W_res, b_res (fp32).
// Output: h_t [B,H] then c_t [B,H], fp32.
// ---------------------------------------------------------------------------

extern "C" void kernel_launch(void* const* d_in, const int* in_sizes, int n_in,
                              void* d_out, int out_size, void* d_ws, size_t ws_size,
                              hipStream_t stream) {
    const float* x  = (const float*)d_in[0];
    const float* h  = (const float*)d_in[1];
    const float* c  = (const float*)d_in[2];
    const float* Wg = (const float*)d_in[3];
    const float* bg = (const float*)d_in[4];
    const float* Wr = (const float*)d_in[5];
    const float* br = (const float*)d_in[6];

    float* h_out = (float*)d_out;
    float* c_out = h_out + (size_t)BATCH * DH;

    // Workspace: bf16 xh, bf16 W_gates^T, bf16 W_res^T.
    unsigned short* xh_bf = (unsigned short*)d_ws;
    unsigned short* wgT   = xh_bf + (size_t)BATCH * DK;
    unsigned short* wrT   = wgT + (size_t)4 * DH * DK;

    {
        size_t n = (size_t)BATCH * DK;
        cvt_xh_kernel<<<dim3((unsigned)((n + 255) / 256)), dim3(256), 0, stream>>>(x, h, xh_bf);
    }
    {
        size_t n = (size_t)4 * DH * DK;
        cvt_wgT_kernel<<<dim3((unsigned)((n + 255) / 256)), dim3(256), 0, stream>>>(Wg, wgT);
    }
    {
        size_t n = (size_t)DH * DIN;
        cvt_wrT_kernel<<<dim3((unsigned)((n + 255) / 256)), dim3(256), 0, stream>>>(Wr, wrT);
    }

    dim3 grid(DH / TN, BATCH / TM);   // (32, 64)
    lstm_fused_wmma_kernel<<<grid, dim3(256), 0, stream>>>(
        xh_bf, wgT, wrT, bg, br, c, h_out, c_out);
}
